// OnnxSTFT_73237782331748
// MI455X (gfx1250) — matrix operand order repaired and checked by reading, not compiled, not run on hardware
//
#include <hip/hip_runtime.h>

typedef __attribute__((ext_vector_type(16))) __bf16 v16bf;
typedef __attribute__((ext_vector_type(8)))  float  v8f;

namespace {
constexpr int kL      = 441000;            // samples per (b,c)
constexpr int kNFFT   = 4096;
constexpr int kHOP    = 1024;
constexpr int kPAD    = kNFFT / 2;         // 2048 reflect pad
constexpr int kNFREQ  = kNFFT / 2 + 1;     // 2049
constexpr int kFRAMES = kL / kHOP + 1;     // 431
constexpr int kBC     = 8;                 // 4 batch * 2 ch
constexpr int kMTILE  = 128;               // freqs per workgroup
constexpr int kNTILE  = 32;                // frames per workgroup
constexpr int kKC     = 32;                // K chunk per WMMA
constexpr int kMP     = ((kNFREQ + kMTILE - 1) / kMTILE) * kMTILE; // 2176
constexpr int kLDSROW = kNTILE + 8;        // 40 bf16 => 80B row stride (conflict-free)
}

// ---- fragment loaders ------------------------------------------------------
// A-matrix 16x32 bf16 (ISA 7.12.2): lanes 0-15 hold K=0..7 (v0-3) and 16..23
// (v4-7); lanes 16-31 hold K=8..15 and 24..31. Per lane: two contiguous runs
// of 8 bf16 (16B each) at +koff and +16+koff.
__device__ inline v16bf loadA(const __bf16* p) {
  union { v16bf v; uint4 q[2]; } u;
  u.q[0] = *(const uint4*)(p);
  u.q[1] = *(const uint4*)(p + 16);
  return u.v;
}
// B-matrix 32x16 bf16: lanes 0-15 hold K=0..7 and 16..23, lanes 16-31 hold
// K=8..15 and 24..31 (two runs of 8 bf16 at +koff and +16+koff).
__device__ inline v16bf loadB(const __bf16* p) {
  union { v16bf v; uint4 q[2]; } u;
  u.q[0] = *(const uint4*)(p);
  u.q[1] = *(const uint4*)(p + 16);
  return u.v;
}

__device__ inline v8f wmma_bf16(v16bf a, v16bf b, v8f c) {
  // (neg_a, A, neg_b, B, c_mod, C, reuse_a, reuse_b)
  return __builtin_amdgcn_wmma_f32_16x16x32_bf16(false, a, false, b, (short)0, c,
                                                 false, false);
}

// ---- pre-pass: filters fp32 -> padded bf16 hi/lo ---------------------------
__global__ __launch_bounds__(256)
void stft_cvt_filters(const float* __restrict__ cosF, const float* __restrict__ sinF,
                      __bf16* __restrict__ cosHi, __bf16* __restrict__ cosLo,
                      __bf16* __restrict__ sinHi, __bf16* __restrict__ sinLo) {
  int idx = blockIdx.x * 256 + threadIdx.x;          // over kMP * kNFFT (row-major)
  if (idx >= kMP * kNFFT) return;
  float c = 0.f, s = 0.f;
  if (idx < kNFREQ * kNFFT) { c = cosF[idx]; s = sinF[idx]; }  // pad rows -> 0
  __bf16 ch = (__bf16)c;
  __bf16 cl = (__bf16)(c - (float)ch);
  __bf16 sh = (__bf16)s;
  __bf16 sl = (__bf16)(s - (float)sh);
  cosHi[idx] = ch; cosLo[idx] = cl; sinHi[idx] = sh; sinLo[idx] = sl;
}

// ---- main GEMM: split-bf16 WMMA STFT ---------------------------------------
__global__ __launch_bounds__(256)
void stft_wmma(const float* __restrict__ x,
               const __bf16* __restrict__ cosHi, const __bf16* __restrict__ cosLo,
               const __bf16* __restrict__ sinHi, const __bf16* __restrict__ sinLo,
               float* __restrict__ out) {
  __shared__ __align__(16) __bf16 ldsHi[kNTILE * kLDSROW];
  __shared__ __align__(16) __bf16 ldsLo[kNTILE * kLDSROW];

  const int lane  = threadIdx.x & 31;
  const int wave  = threadIdx.x >> 5;
  const int tBase = blockIdx.x * kNTILE;             // frame tile
  const int mBase = blockIdx.y * kMTILE + wave * 16; // 16 freq rows per wave
  const int bc    = blockIdx.z;

  // A fragment addressing (per ISA lane layout)
  const int arow  = mBase + (lane & 15);
  const int akoff = (lane >> 4) * 8;
  const __bf16* cHrow = cosHi + (size_t)arow * kNFFT + akoff;
  const __bf16* cLrow = cosLo + (size_t)arow * kNFFT + akoff;
  const __bf16* sHrow = sinHi + (size_t)arow * kNFFT + akoff;
  const __bf16* sLrow = sinLo + (size_t)arow * kNFFT + akoff;

  const float* xb = x + (size_t)bc * kL;

  // B fragment LDS offset: row = frame (lane&15), col base = (lane>=16)*8
  const int bro = (lane & 15) * kLDSROW + (lane >> 4) * 8;

  v8f accRe0 = {}; v8f accIm0 = {};
  v8f accRe1 = {}; v8f accIm1 = {};

  for (int k0 = 0; k0 < kNFFT; k0 += kKC) {
    __syncthreads();
    // Stage B: 32 frames x 32 taps, reflect-pad + fp32->bf16 hi/lo split.
    for (int e = threadIdx.x; e < kNTILE * kKC; e += 256) {
      int t = e >> 5, k = e & 31;
      int i = (tBase + t) * kHOP + k0 + k - kPAD;
      i = (i < 0) ? -i : i;
      i = (i >= kL) ? (2 * kL - 2 - i) : i;
      float v = xb[i];
      __bf16 h = (__bf16)v;
      ldsHi[t * kLDSROW + k] = h;
      ldsLo[t * kLDSROW + k] = (__bf16)(v - (float)h);
    }
    __syncthreads();

    // Prefetch next filter chunk (L2-resident: 67MB filters < 192MB L2).
    int kpf = k0 + 4 * kKC;
    if (kpf < kNFFT) {
      __builtin_prefetch(cHrow + kpf, 0, 1);
      __builtin_prefetch(sHrow + kpf, 0, 1);
    }

    // A fragments (cos/sin, hi/lo)
    v16bf aCH = loadA(cHrow + k0);
    v16bf aCL = loadA(cLrow + k0);
    v16bf aSH = loadA(sHrow + k0);
    v16bf aSL = loadA(sLrow + k0);

    // B fragments (two N-subtiles of 16 frames)
    v16bf bH0 = loadB(ldsHi + bro);
    v16bf bL0 = loadB(ldsLo + bro);
    v16bf bH1 = loadB(ldsHi + 16 * kLDSROW + bro);
    v16bf bL1 = loadB(ldsLo + 16 * kLDSROW + bro);

    // 3-term split-precision accumulate (lo*lo term below fp32 noise floor)
    accRe0 = wmma_bf16(aCH, bH0, accRe0);
    accRe0 = wmma_bf16(aCH, bL0, accRe0);
    accRe0 = wmma_bf16(aCL, bH0, accRe0);
    accIm0 = wmma_bf16(aSH, bH0, accIm0);
    accIm0 = wmma_bf16(aSH, bL0, accIm0);
    accIm0 = wmma_bf16(aSL, bH0, accIm0);

    accRe1 = wmma_bf16(aCH, bH1, accRe1);
    accRe1 = wmma_bf16(aCH, bL1, accRe1);
    accRe1 = wmma_bf16(aCL, bH1, accRe1);
    accIm1 = wmma_bf16(aSH, bH1, accIm1);
    accIm1 = wmma_bf16(aSH, bL1, accIm1);
    accIm1 = wmma_bf16(aSL, bH1, accIm1);
  }

  // Store: C/D layout VGPR r -> M = r + 8*(lane>=16), N = lane&15.
  // Output (b,c,freq,time,2): write contiguous {re,im} as float2.
#pragma unroll
  for (int j = 0; j < 2; ++j) {
    v8f re = j ? accRe1 : accRe0;
    v8f im = j ? accIm1 : accIm0;
    int t = tBase + j * 16 + (lane & 15);
    if (t < kFRAMES) {
#pragma unroll
      for (int r = 0; r < 8; ++r) {
        int freq = mBase + r + (lane >> 4) * 8;
        if (freq < kNFREQ) {
          float2 v; v.x = re[r]; v.y = im[r];
          *(float2*)(out + (((size_t)bc * kNFREQ + freq) * kFRAMES + t) * 2) = v;
        }
      }
    }
  }
}

extern "C" void kernel_launch(void* const* d_in, const int* in_sizes, int n_in,
                              void* d_out, int out_size, void* d_ws, size_t ws_size,
                              hipStream_t stream) {
  const float* x    = (const float*)d_in[0];
  const float* cosF = (const float*)d_in[1];
  const float* sinF = (const float*)d_in[2];
  float* out = (float*)d_out;

  const size_t fElems = (size_t)kMP * kNFFT;         // 8,912,896 per array
  __bf16* cosHi = (__bf16*)d_ws;
  __bf16* cosLo = cosHi + fElems;
  __bf16* sinHi = cosLo + fElems;
  __bf16* sinLo = sinHi + fElems;                    // total ~71.3 MB workspace

  const int cvtBlocks = (int)((fElems + 255) / 256);
  stft_cvt_filters<<<cvtBlocks, 256, 0, stream>>>(cosF, sinF,
                                                  cosHi, cosLo, sinHi, sinLo);

  dim3 grid((kFRAMES + kNTILE - 1) / kNTILE,         // 14 frame tiles
            kMP / kMTILE,                            // 17 freq tiles
            kBC);                                    // 8 batch-channels
  stft_wmma<<<grid, 256, 0, stream>>>(x, cosHi, cosLo, sinHi, sinLo, out);
}
